// LSTM_78477642432860
// MI455X (gfx1250) — matrix operand order repaired and checked by reading, not compiled, not run on hardware
//
#include <hip/hip_runtime.h>
#include <hip/hip_bf16.h>

typedef unsigned short u16;
typedef __attribute__((ext_vector_type(16))) __bf16 v16bf;
typedef __attribute__((ext_vector_type(8)))  float  v8f;

#define B_  256
#define T_  256
#define I_  512
#define H_  1024
#define G4_ 4096
#define KC_ 1536   // I + H combined K

// ---- f32 -> bf16, round to nearest even ----
__device__ __forceinline__ u16 f2bf(float f) {
  unsigned int u = __float_as_uint(f);
  u += 0x7FFFu + ((u >> 16) & 1u);
  return (u16)(u >> 16);
}

// load 16 contiguous bf16 (32 bytes, 32B-aligned)
__device__ __forceinline__ v16bf load16(const u16* p) {
  union { uint4 q[2]; v16bf v; } u;
  u.q[0] = *reinterpret_cast<const uint4*>(p);
  u.q[1] = *reinterpret_cast<const uint4*>(p + 8);
  return u.v;
}
// A fragment: two disjoint 8-element (16B) chunks (adjacent at call sites: k = 16*lhalf + s, as B)
__device__ __forceinline__ v16bf load8x2(const u16* p0, const u16* p1) {
  union { uint4 q[2]; v16bf v; } u;
  u.q[0] = *reinterpret_cast<const uint4*>(p0);
  u.q[1] = *reinterpret_cast<const uint4*>(p1);
  return u.v;
}

#define WMMA_BF16(ACC, A, Bf) \
  ACC = __builtin_amdgcn_wmma_f32_16x16x32_bf16(false, (A), false, (Bf), (short)0, (ACC), false, false)

// ---------------- prep: combined bf16 weights Wcomb[4H][1536] ----------------
__global__ void prep_weights(const float* __restrict__ Wfx, const float* __restrict__ Wfh,
                             const float* __restrict__ Wix, const float* __restrict__ Wih,
                             const float* __restrict__ Wox, const float* __restrict__ Woh,
                             const float* __restrict__ Wcx, const float* __restrict__ Wch,
                             u16* __restrict__ Wcomb) {
  int n = blockIdx.x;            // 0..4095
  int g = n >> 10;
  int r = n & 1023;
  const float* wx; const float* wh;
  if (g == 0)      { wx = Wfx; wh = Wfh; }
  else if (g == 1) { wx = Wix; wh = Wih; }
  else if (g == 2) { wx = Wox; wh = Woh; }
  else             { wx = Wcx; wh = Wch; }
  wx += (size_t)r * I_;
  wh += (size_t)r * H_;
  u16* out = Wcomb + (size_t)n * KC_;
  for (int k = threadIdx.x; k < KC_; k += blockDim.x) {
    float v = (k < I_) ? wx[k] : wh[k - I_];
    out[k] = f2bf(v);
  }
}

// ---------------- prep: x [B][T][I] f32 -> xbf [T][B][I] bf16 ----------------
__global__ void prep_x(const float* __restrict__ x, u16* __restrict__ xbf) {
  size_t total = (size_t)B_ * T_ * I_;
  for (size_t idx = (size_t)blockIdx.x * blockDim.x + threadIdx.x;
       idx < total; idx += (size_t)gridDim.x * blockDim.x) {
    size_t b = idx / ((size_t)T_ * I_);
    size_t rem = idx - b * ((size_t)T_ * I_);
    size_t t = rem / I_;
    size_t i = rem - t * I_;
    xbf[(t * B_ + b) * I_ + i] = f2bf(x[idx]);
  }
}

// ---------------- init: zero c, hA, hfinal ----------------
__global__ void init_state(float* __restrict__ c, u16* __restrict__ hA,
                           float* __restrict__ hf) {
  int idx = blockIdx.x * blockDim.x + threadIdx.x;
  if (idx < B_ * H_) { c[idx] = 0.0f; hA[idx] = 0; hf[idx] = 0.0f; }
}

// ---------------- one LSTM timestep ----------------
// grid: 64 j-tiles; block: 256 threads = 8 waves (wave32).
// Wave w owns TWO 16x16 M-tiles (rows m0=w*32 .. +31) for column j0, all 4
// gates => 8 WMMAs per K-step with only 4 A + 8 B b128 loads (B fragments
// reused across both M-tiles). Cell update is lane-local.
__global__ __launch_bounds__(256)
void lstm_step(const u16* __restrict__ xbf_t,   // [B][I] bf16 for this t
               const u16* __restrict__ h_in,    // [B][H] bf16
               const u16* __restrict__ Wcomb,   // [4H][KC] bf16
               const float* __restrict__ bfv, const float* __restrict__ biv,
               const float* __restrict__ bov, const float* __restrict__ bcv,
               float* __restrict__ c,           // [B][H] f32 (in/out)
               u16* __restrict__ h_out,         // [B][H] bf16
               float* __restrict__ h_f32)       // [B][H] f32
{
  const int wave  = threadIdx.x >> 5;
  const int lane  = threadIdx.x & 31;
  const int lhalf = lane >> 4;        // 0 / 1
  const int lmod  = lane & 15;

  const int m0 = wave * 32;           // two M-tiles: m0, m0+16
  const int j0 = blockIdx.x * 16;

  // A addressing: rows m0+lmod and m0+16+lmod
  const u16* xrow0 = xbf_t + (size_t)(m0 + lmod) * I_;
  const u16* xrow1 = xbf_t + (size_t)(m0 + 16 + lmod) * I_;
  const u16* hrow0 = h_in  + (size_t)(m0 + lmod) * H_;
  const u16* hrow1 = h_in  + (size_t)(m0 + 16 + lmod) * H_;

  // B addressing: column n = lmod within tile; row of Wcomb is K-major.
  const u16* brow0 = Wcomb + ((size_t)(0 * H_ + j0 + lmod)) * KC_ + lhalf * 16;
  const u16* brow1 = Wcomb + ((size_t)(1 * H_ + j0 + lmod)) * KC_ + lhalf * 16;
  const u16* brow2 = Wcomb + ((size_t)(2 * H_ + j0 + lmod)) * KC_ + lhalf * 16;
  const u16* brow3 = Wcomb + ((size_t)(3 * H_ + j0 + lmod)) * KC_ + lhalf * 16;

  v8f zero = {0.f,0.f,0.f,0.f,0.f,0.f,0.f,0.f};
  v8f f0 = zero, i0 = zero, o0 = zero, c0 = zero;   // M-tile 0
  v8f f1 = zero, i1 = zero, o1 = zero, c1 = zero;   // M-tile 1

  const int aoff = lhalf * 16;

  // ---- phase 1: K over x projection (KC cols [0, 512)) ----
  for (int kk = 0; kk < I_; kk += 32) {
    v16bf a0 = load8x2(xrow0 + kk + aoff, xrow0 + kk + 8 + aoff);
    v16bf a1 = load8x2(xrow1 + kk + aoff, xrow1 + kk + 8 + aoff);
    v16bf b0 = load16(brow0 + kk);
    v16bf b1 = load16(brow1 + kk);
    v16bf b2 = load16(brow2 + kk);
    v16bf b3 = load16(brow3 + kk);
    WMMA_BF16(f0, a0, b0); WMMA_BF16(f1, a1, b0);
    WMMA_BF16(i0, a0, b1); WMMA_BF16(i1, a1, b1);
    WMMA_BF16(o0, a0, b2); WMMA_BF16(o1, a1, b2);
    WMMA_BF16(c0, a0, b3); WMMA_BF16(c1, a1, b3);
  }
  // ---- phase 2: K over h projection (KC cols [512, 1536)) ----
  for (int kk = 0; kk < H_; kk += 32) {
    const int kb = I_ + kk;
    v16bf a0 = load8x2(hrow0 + kk + aoff, hrow0 + kk + 8 + aoff);
    v16bf a1 = load8x2(hrow1 + kk + aoff, hrow1 + kk + 8 + aoff);
    v16bf b0 = load16(brow0 + kb);
    v16bf b1 = load16(brow1 + kb);
    v16bf b2 = load16(brow2 + kb);
    v16bf b3 = load16(brow3 + kb);
    WMMA_BF16(f0, a0, b0); WMMA_BF16(f1, a1, b0);
    WMMA_BF16(i0, a0, b1); WMMA_BF16(i1, a1, b1);
    WMMA_BF16(o0, a0, b2); WMMA_BF16(o1, a1, b2);
    WMMA_BF16(c0, a0, b3); WMMA_BF16(c1, a1, b3);
  }

  // ---- cell update (lane-local): C/D layout: VGPR r -> M = r + lhalf*8, N = lmod ----
  const int j = j0 + lmod;
  const float bF = bfv[j], bI = biv[j], bO = bov[j], bC = bcv[j];
  #pragma unroll
  for (int tile = 0; tile < 2; ++tile) {
    const v8f& af = tile ? f1 : f0;
    const v8f& ai = tile ? i1 : i0;
    const v8f& ao = tile ? o1 : o0;
    const v8f& ac = tile ? c1 : c0;
    #pragma unroll
    for (int r = 0; r < 8; ++r) {
      const int mr = m0 + tile * 16 + r + lhalf * 8;
      const size_t idx = (size_t)mr * H_ + j;
      float pf = af[r] + bF;
      float pi = ai[r] + bI;
      float po = ao[r] + bO;
      float pc = ac[r] + bC;
      float fg = 1.0f / (1.0f + __expf(-pf));
      float ig = 1.0f / (1.0f + __expf(-pi));
      float og = 1.0f / (1.0f + __expf(-po));
      float ct = tanhf(pc);
      float cn = fg * c[idx] + ig * ct;
      float hn = og * tanhf(cn);
      c[idx]     = cn;
      h_out[idx] = f2bf(hn);
      h_f32[idx] = hn;
    }
  }
}

// ---------------- head: relu -> [64] relu -> [10] softmax ----------------
__global__ void head_kernel(const float* __restrict__ h,
                            const float* __restrict__ W1, const float* __restrict__ b1,
                            const float* __restrict__ W2, const float* __restrict__ b2,
                            float* __restrict__ out) {
  __shared__ float h1[64];
  __shared__ float lg[10];
  const int row = blockIdx.x;
  const int u = threadIdx.x;           // 64 threads
  const float* hr = h + (size_t)row * H_;
  {
    const float* w = W1 + (size_t)u * H_;
    float s = b1[u];
    for (int k = 0; k < H_; ++k) s += fmaxf(hr[k], 0.0f) * w[k];
    h1[u] = fmaxf(s, 0.0f);
  }
  __syncthreads();
  if (u < 10) {
    const float* w = W2 + (size_t)u * 64;
    float s = b2[u];
    for (int k = 0; k < 64; ++k) s += h1[k] * w[k];
    lg[u] = s;
  }
  __syncthreads();
  if (u < 10) {
    float mx = lg[0];
    for (int t = 1; t < 10; ++t) mx = fmaxf(mx, lg[t]);
    float den = 0.0f;
    for (int t = 0; t < 10; ++t) den += __expf(lg[t] - mx);
    out[(size_t)row * 10 + u] = __expf(lg[u] - mx) / den;
  }
}

extern "C" void kernel_launch(void* const* d_in, const int* in_sizes, int n_in,
                              void* d_out, int out_size, void* d_ws, size_t ws_size,
                              hipStream_t stream) {
  const float* x   = (const float*)d_in[0];
  const float* Wfx = (const float*)d_in[1];
  const float* Wfh = (const float*)d_in[2];
  const float* bf  = (const float*)d_in[3];
  const float* Wix = (const float*)d_in[4];
  const float* Wih = (const float*)d_in[5];
  const float* bi  = (const float*)d_in[6];
  const float* Wox = (const float*)d_in[7];
  const float* Woh = (const float*)d_in[8];
  const float* bo  = (const float*)d_in[9];
  const float* Wcx = (const float*)d_in[10];
  const float* Wch = (const float*)d_in[11];
  const float* bc  = (const float*)d_in[12];
  const float* W1  = (const float*)d_in[13];
  const float* b1  = (const float*)d_in[14];
  const float* W2  = (const float*)d_in[15];
  const float* b2  = (const float*)d_in[16];
  float* out = (float*)d_out;

  char* ws = (char*)d_ws;
  size_t off = 0;
  u16*   Wcomb = (u16*)(ws + off);   off += (size_t)G4_ * KC_ * sizeof(u16);      // 12 MiB
  u16*   xbf   = (u16*)(ws + off);   off += (size_t)T_ * B_ * I_ * sizeof(u16);   // 64 MiB
  float* cbuf  = (float*)(ws + off); off += (size_t)B_ * H_ * sizeof(float);      // 1 MiB
  u16*   hA    = (u16*)(ws + off);   off += (size_t)B_ * H_ * sizeof(u16);        // 0.5 MiB
  u16*   hB    = (u16*)(ws + off);   off += (size_t)B_ * H_ * sizeof(u16);        // 0.5 MiB
  float* hf    = (float*)(ws + off); off += (size_t)B_ * H_ * sizeof(float);      // 1 MiB

  prep_weights<<<dim3(G4_), dim3(256), 0, stream>>>(Wfx, Wfh, Wix, Wih, Wox, Woh,
                                                    Wcx, Wch, Wcomb);
  prep_x<<<dim3(4096), dim3(256), 0, stream>>>(x, xbf);
  init_state<<<dim3((B_ * H_ + 255) / 256), dim3(256), 0, stream>>>(cbuf, hA, hf);

  for (int t = 0; t < T_; ++t) {
    const u16* xt   = xbf + (size_t)t * B_ * I_;
    const u16* hin  = (t & 1) ? hB : hA;
    u16*       hout = (t & 1) ? hA : hB;
    lstm_step<<<dim3(64), dim3(256), 0, stream>>>(xt, hin, Wcomb,
                                                  bf, bi, bo, bc,
                                                  cbuf, hout, hf);
  }

  head_kernel<<<dim3(B_), dim3(64), 0, stream>>>(hf, W1, b1, W2, b2, out);
}
